// BayesBlock_8650064134166
// MI455X (gfx1250) — matrix operand order repaired and checked by reading, not compiled, not run on hardware
//
#include <hip/hip_runtime.h>
#include <hip/hip_bf16.h>

// ---------------------------------------------------------------------------
// Problem constants (match reference)
// ---------------------------------------------------------------------------
static constexpr int Dd = 3;     // DEPTH
static constexpr int Ss = 4;     // N_SMPL
static constexpr int Ff = 2048;  // F
static constexpr int Bb = 4096;  // B

typedef __bf16 v16bf __attribute__((ext_vector_type(16)));
typedef __bf16 v8bf  __attribute__((ext_vector_type(8)));
typedef __bf16 v4bf  __attribute__((ext_vector_type(4)));
typedef float  v8f   __attribute__((ext_vector_type(8)));

union Frag16 { v16bf v; v8bf h[2]; };

__device__ __forceinline__ float softplusf(float r) {
    return (r > 20.0f) ? r : log1pf(expf(r));
}

__device__ __forceinline__ void split_bf16(float v, __bf16& hi, __bf16& lo) {
    hi = (__bf16)v;
    lo = (__bf16)(v - (float)hi);
}

// gfx1250 async global->LDS copy (ASYNCcnt-tracked, bypasses VGPRs).
// vdst = per-lane LDS byte address, vaddr = per-lane 64-bit global address.
__device__ __forceinline__ void async_copy_b128(const void* g, void* l) {
    unsigned lds         = (unsigned)(unsigned long long)l;   // low 32 bits = LDS offset
    unsigned long long ga = (unsigned long long)g;
    asm volatile("global_load_async_to_lds_b128 %0, %1, off"
                 :: "v"(lds), "v"(ga) : "memory");
}
#define WAIT_ASYNCCNT(n) asm volatile("s_wait_asynccnt " #n ::: "memory")

// ---------------------------------------------------------------------------
// Kernel: split fp32 array into bf16 hi/lo pair
// ---------------------------------------------------------------------------
__global__ __launch_bounds__(256)
void k_split(const float* __restrict__ src, __bf16* __restrict__ hi,
             __bf16* __restrict__ lo, long n) {
    long i = ((long)blockIdx.x * 256 + threadIdx.x) * 4;
    if (i >= n) return;
    float4 v = *(const float4*)(src + i);
    float vv[4] = {v.x, v.y, v.z, v.w};
    v4bf h, l;
#pragma unroll
    for (int j = 0; j < 4; ++j) {
        __bf16 hh, ll;
        split_bf16(vv[j], hh, ll);
        h[j] = hh; l[j] = ll;
    }
    *(v4bf*)(hi + i) = h;
    *(v4bf*)(lo + i) = l;
}

// ---------------------------------------------------------------------------
// Kernel: perturbed weights  W[d,s] = w_mu[d] + softplus(w_rho[d]) * eps_w[d,s]
// stored split into bf16 hi/lo, layout [D,S,F,F]
// ---------------------------------------------------------------------------
__global__ __launch_bounds__(256)
void k_perturb_w(const float* __restrict__ w_mu, const float* __restrict__ w_rho,
                 const float* __restrict__ eps_w,
                 __bf16* __restrict__ Whi, __bf16* __restrict__ Wlo) {
    const long ff = (long)Ff * Ff;
    long i = ((long)blockIdx.x * 256 + threadIdx.x) * 4;
    long ds  = i / ff;          // 0..D*S-1
    long rem = i - ds * ff;
    long d   = ds / Ss;
    float4 mu  = *(const float4*)(w_mu  + d * ff + rem);
    float4 rho = *(const float4*)(w_rho + d * ff + rem);
    float4 ep  = *(const float4*)(eps_w + ds * ff + rem);
    float m[4] = {mu.x, mu.y, mu.z, mu.w};
    float r[4] = {rho.x, rho.y, rho.z, rho.w};
    float e[4] = {ep.x, ep.y, ep.z, ep.w};
    v4bf h, l;
#pragma unroll
    for (int j = 0; j < 4; ++j) {
        float w = fmaf(softplusf(r[j]), e[j], m[j]);
        __bf16 hh, ll;
        split_bf16(w, hh, ll);
        h[j] = hh; l[j] = ll;
    }
    *(v4bf*)(Whi + ds * ff + rem) = h;
    *(v4bf*)(Wlo + ds * ff + rem) = l;
}

// ---------------------------------------------------------------------------
// Kernel: perturbed biases b[d,s] = b_mu[d] + softplus(b_rho[d]) * eps_b[d,s]
// ---------------------------------------------------------------------------
__global__ __launch_bounds__(256)
void k_perturb_b(const float* __restrict__ b_mu, const float* __restrict__ b_rho,
                 const float* __restrict__ eps_b, float* __restrict__ bp) {
    int i = blockIdx.x * 256 + threadIdx.x;     // < D*S*F
    int ds  = i / Ff;
    int o   = i - ds * Ff;
    int d   = ds / Ss;
    float m = b_mu[d * Ff + o];
    float r = b_rho[d * Ff + o];
    float e = eps_b[ds * Ff + o];
    bp[i] = fmaf(softplusf(r), e, m);
}

// ---------------------------------------------------------------------------
// GEMM: O[s,b,o] = relu( sum_i A[s,b,i] * W[s,o,i] + bias[s,o] )
// bf16x3 compensated WMMA, fp32 accum. Block tile 128x64, K tile 32,
// 256 threads = 8 wave32 (4x2), wave tile 32x32.
// Double-buffered LDS staged via global_load_async_to_lds_b128 (ASYNCcnt).
// ---------------------------------------------------------------------------
static constexpr int BM = 128, BN = 64, BK = 32, LPAD = 8;
static constexpr int LSTR = BK + LPAD;  // LDS row stride (bf16) = 40 -> 80B, 16B aligned
static constexpr int KT   = Ff / BK;    // 64 K-steps

__global__ __launch_bounds__(256)
void k_gemm(const __bf16* __restrict__ Ahi, const __bf16* __restrict__ Alo,
            const __bf16* __restrict__ Whi, const __bf16* __restrict__ Wlo,
            const float*  __restrict__ bias,     // [S,F] for this layer
            __bf16* __restrict__ Ohi, __bf16* __restrict__ Olo,
            long aStrideS)                       // 0 (layer 0 broadcast) or B*F
{
    // [stage][hi/lo][row][col]
    __shared__ __attribute__((aligned(16))) __bf16 sA[2][2][BM][LSTR]; // 40 KB
    __shared__ __attribute__((aligned(16))) __bf16 sB[2][2][BN][LSTR]; // 20 KB

    const int tid  = threadIdx.x;
    const int lane = tid & 31;
    const int wid  = tid >> 5;
    const int waveM = (wid & 3) * 32;   // 4 waves along M
    const int waveN = (wid >> 2) * 32;  // 2 waves along N
    const int s = blockIdx.z;
    const long mBlock = (long)blockIdx.x * BM;
    const long nBlock = (long)blockIdx.y * BN;

    const __bf16* As_hi = Ahi + (long)s * aStrideS;
    const __bf16* As_lo = Alo + (long)s * aStrideS;
    const __bf16* Ws_hi = Whi + (long)s * Ff * Ff;
    const __bf16* Ws_lo = Wlo + (long)s * Ff * Ff;

    // Issue one stage of async copies: 6 b128 ops per thread.
    auto stage = [&](int st, int k0) {
#pragma unroll
        for (int cc = 0; cc < 2; ++cc) {            // A: 512 chunks x (hi,lo)
            int c   = tid + cc * 256;
            int row = c >> 2;
            int col = (c & 3) * 8;
            long off = (mBlock + row) * (long)Ff + k0 + col;
            async_copy_b128(As_hi + off, &sA[st][0][row][col]);
            async_copy_b128(As_lo + off, &sA[st][1][row][col]);
        }
        {                                           // B: 256 chunks x (hi,lo)
            int row = tid >> 2;
            int col = (tid & 3) * 8;
            long off = (nBlock + row) * (long)Ff + k0 + col;
            async_copy_b128(Ws_hi + off, &sB[st][0][row][col]);
            async_copy_b128(Ws_lo + off, &sB[st][1][row][col]);
        }
    };

    const v8f vzero = {0.f, 0.f, 0.f, 0.f, 0.f, 0.f, 0.f, 0.f};
    v8f acc[2][2];
#pragma unroll
    for (int m = 0; m < 2; ++m)
#pragma unroll
        for (int n = 0; n < 2; ++n) acc[m][n] = vzero;

    stage(0, 0);   // prologue: tile 0 in flight

    for (int kt = 0; kt < KT; ++kt) {
        const int st = kt & 1;
        if (kt + 1 < KT) {
            stage(st ^ 1, (kt + 1) * BK);   // keep next tile in flight
            WAIT_ASYNCCNT(6);               // oldest 6 (tile kt) complete, in order
        } else {
            WAIT_ASYNCCNT(0);
        }
        __syncthreads();                    // tile kt visible to all waves

        // A fragments (16-bit A 16x32: lanes<16 K=0..7,16..23; lanes>=16 K=8..15,24..31)
        Frag16 aHi[2], aLo[2];
#pragma unroll
        for (int m = 0; m < 2; ++m) {
            int mr = waveM + m * 16 + (lane & 15);
            int ka = (lane < 16) ? 0 : 8;
            aHi[m].h[0] = *(const v8bf*)&sA[st][0][mr][ka];
            aHi[m].h[1] = *(const v8bf*)&sA[st][0][mr][ka + 16];
            aLo[m].h[0] = *(const v8bf*)&sA[st][1][mr][ka];
            aLo[m].h[1] = *(const v8bf*)&sA[st][1][mr][ka + 16];
        }
        // B fragments (32x16: lane = N; lanes<16 K=0..7,16..23; lanes>=16 K=8..15,24..31)
#pragma unroll
        for (int n = 0; n < 2; ++n) {
            int nr = waveN + n * 16 + (lane & 15);
            int kb = (lane < 16) ? 0 : 8;
            Frag16 bHi, bLo;
            bHi.h[0] = *(const v8bf*)&sB[st][0][nr][kb];
            bHi.h[1] = *(const v8bf*)&sB[st][0][nr][kb + 16];
            bLo.h[0] = *(const v8bf*)&sB[st][1][nr][kb];
            bLo.h[1] = *(const v8bf*)&sB[st][1][nr][kb + 16];
#pragma unroll
            for (int m = 0; m < 2; ++m) {
                // bf16x3: hi*hi + hi*lo + lo*hi  (drop lo*lo)
                acc[m][n] = __builtin_amdgcn_wmma_f32_16x16x32_bf16(
                    false, aHi[m].v, false, bHi.v, (short)0, acc[m][n], false, false);
                acc[m][n] = __builtin_amdgcn_wmma_f32_16x16x32_bf16(
                    false, aHi[m].v, false, bLo.v, (short)0, acc[m][n], false, false);
                acc[m][n] = __builtin_amdgcn_wmma_f32_16x16x32_bf16(
                    false, aLo[m].v, false, bHi.v, (short)0, acc[m][n], false, false);
            }
        }
        __syncthreads();   // all waves done reading stage st before it is re-filled
    }

    // Epilogue: bias + relu + bf16 split store.
    const float* bS = bias + (long)s * Ff;
#pragma unroll
    for (int m = 0; m < 2; ++m) {
        long mbase = mBlock + waveM + m * 16 + ((lane >> 4) << 3);
#pragma unroll
        for (int n = 0; n < 2; ++n) {
            long gn = nBlock + waveN + n * 16 + (lane & 15);
            float bv = bS[gn];
#pragma unroll
            for (int r = 0; r < 8; ++r) {
                float v = fmaxf(acc[m][n][r] + bv, 0.0f);
                __bf16 h, l;
                split_bf16(v, h, l);
                long oi = ((long)s * Bb + (mbase + r)) * (long)Ff + gn;
                Ohi[oi] = h;
                Olo[oi] = l;
            }
        }
    }
}

// ---------------------------------------------------------------------------
// Block reductions (wave32 shuffles + LDS across 8 waves)
// ---------------------------------------------------------------------------
__device__ __forceinline__ float blockReduceSum(float v, float* sred, int tid) {
#pragma unroll
    for (int m = 16; m >= 1; m >>= 1) v += __shfl_xor(v, m, 32);
    if ((tid & 31) == 0) sred[tid >> 5] = v;
    __syncthreads();
    float t = sred[0];
#pragma unroll
    for (int w = 1; w < 8; ++w) t += sred[w];
    __syncthreads();
    return t;
}

__device__ __forceinline__ float blockReduceMax(float v, float* sred, int tid) {
#pragma unroll
    for (int m = 16; m >= 1; m >>= 1) v = fmaxf(v, __shfl_xor(v, m, 32));
    if ((tid & 31) == 0) sred[tid >> 5] = v;
    __syncthreads();
    float t = sred[0];
#pragma unroll
    for (int w = 1; w < 8; ++w) t = fmaxf(t, sred[w]);
    __syncthreads();
    return t;
}

// ---------------------------------------------------------------------------
// Sparsemax (sort-free bisection for tau) + mean over samples + residual.
// One block per batch row b; 256 threads, 8 columns each.
// ---------------------------------------------------------------------------
__global__ __launch_bounds__(256)
void k_sparsemax(const __bf16* __restrict__ Hhi, const __bf16* __restrict__ Hlo,
                 const float* __restrict__ x, float* __restrict__ out) {
    __shared__ float sred[8];
    const int b = blockIdx.x;
    const int tid = threadIdx.x;
    float accum[8] = {0, 0, 0, 0, 0, 0, 0, 0};

    for (int s = 0; s < Ss; ++s) {
        const long base = ((long)s * Bb + b) * (long)Ff;
        float z[8];
        float lmax = -3.0e38f;
#pragma unroll
        for (int j = 0; j < 8; ++j) {
            int col = tid + j * 256;
            float v = (float)Hhi[base + col] + (float)Hlo[base + col];
            z[j] = v;
            lmax = fmaxf(lmax, v);
        }
        float rmax = blockReduceMax(lmax, sred, tid);

        // g(tau) = sum relu(z - tau) - 1 strictly decreasing; root in [rmax-1, rmax].
        float lo = rmax - 1.0f, hi = rmax, tau = rmax - 0.5f;
        for (int it = 0; it < 32; ++it) {
            tau = 0.5f * (lo + hi);
            float ls = 0.0f;
#pragma unroll
            for (int j = 0; j < 8; ++j) ls += fmaxf(z[j] - tau, 0.0f);
            float g = blockReduceSum(ls, sred, tid);
            if (g >= 1.0f) lo = tau; else hi = tau;
        }
        tau = 0.5f * (lo + hi);
#pragma unroll
        for (int j = 0; j < 8; ++j) accum[j] += fmaxf(z[j] - tau, 0.0f);
    }

    const long xb = (long)b * Ff;
#pragma unroll
    for (int j = 0; j < 8; ++j) {
        int col = tid + j * 256;
        out[xb + col] = 0.25f * accum[j] + x[xb + col];
    }
}

// ---------------------------------------------------------------------------
// Host-side orchestration
// ---------------------------------------------------------------------------
extern "C" void kernel_launch(void* const* d_in, const int* in_sizes, int n_in,
                              void* d_out, int out_size, void* d_ws, size_t ws_size,
                              hipStream_t stream) {
    (void)in_sizes; (void)n_in; (void)out_size; (void)ws_size;

    const float* x     = (const float*)d_in[0];
    const float* w_mu  = (const float*)d_in[1];
    const float* w_rho = (const float*)d_in[2];
    const float* b_mu  = (const float*)d_in[3];
    const float* b_rho = (const float*)d_in[4];
    const float* eps_w = (const float*)d_in[5];
    const float* eps_b = (const float*)d_in[6];
    float* out = (float*)d_out;

    const long ff  = (long)Ff * Ff;
    const long bf  = (long)Bb * Ff;
    const long sbf = (long)Ss * bf;

    char* ws = (char*)d_ws;
    size_t off = 0;
    auto take = [&](size_t bytes) -> char* {
        char* p = ws + off;
        off += (bytes + 255) & ~(size_t)255;
        return p;
    };
    __bf16* Whi  = (__bf16*)take((size_t)Dd * Ss * ff * 2);
    __bf16* Wlo  = (__bf16*)take((size_t)Dd * Ss * ff * 2);
    float*  bp   = (float*)take((size_t)Dd * Ss * Ff * 4);
    __bf16* xhi  = (__bf16*)take((size_t)bf * 2);
    __bf16* xlo  = (__bf16*)take((size_t)bf * 2);
    __bf16* h0hi = (__bf16*)take((size_t)sbf * 2);
    __bf16* h0lo = (__bf16*)take((size_t)sbf * 2);
    __bf16* h1hi = (__bf16*)take((size_t)sbf * 2);
    __bf16* h1lo = (__bf16*)take((size_t)sbf * 2);

    // 1) split x into bf16 hi/lo
    k_split<<<(unsigned)(bf / (256 * 4)), 256, 0, stream>>>(x, xhi, xlo, bf);

    // 2) perturbed weights (all layers/samples) split into bf16 hi/lo
    k_perturb_w<<<(unsigned)((long)Dd * Ss * ff / (256 * 4)), 256, 0, stream>>>(
        w_mu, w_rho, eps_w, Whi, Wlo);

    // 3) perturbed biases
    k_perturb_b<<<(Dd * Ss * Ff) / 256, 256, 0, stream>>>(b_mu, b_rho, eps_b, bp);

    // 4) three WMMA GEMM layers (bias+relu fused)
    dim3 gg(Bb / BM, Ff / BN, Ss);
    dim3 gb(256);
    // layer 0: A = x broadcast over samples (stride 0)
    k_gemm<<<gg, gb, 0, stream>>>(xhi, xlo, Whi + 0 * Ss * ff, Wlo + 0 * Ss * ff,
                                  bp + 0 * Ss * Ff, h0hi, h0lo, (long)0);
    // layer 1
    k_gemm<<<gg, gb, 0, stream>>>(h0hi, h0lo, Whi + 1 * Ss * ff, Wlo + 1 * Ss * ff,
                                  bp + 1 * Ss * Ff, h1hi, h1lo, bf);
    // layer 2 (reuse h0 buffers as output)
    k_gemm<<<gg, gb, 0, stream>>>(h1hi, h1lo, Whi + 2 * Ss * ff, Wlo + 2 * Ss * ff,
                                  bp + 2 * Ss * Ff, h0hi, h0lo, bf);

    // 5) sparsemax per (s,b) row, mean over samples, residual add
    k_sparsemax<<<Bb, 256, 0, stream>>>(h0hi, h0lo, x, out);
}
